// DAMLSTM_42090679500991
// MI455X (gfx1250) — matrix operand order repaired and checked by reading, not compiled, not run on hardware
//
#include <hip/hip_runtime.h>
#include <hip/hip_bf16.h>

// ---------------- CDNA5 WMMA types ----------------
typedef __attribute__((ext_vector_type(16))) __bf16 v16bf;
typedef __attribute__((ext_vector_type(8)))  __bf16 v8bf;
typedef __attribute__((ext_vector_type(8)))  float  v8f;

#define B_ 64
#define T_ 256
#define F_ 1024
#define H_ 1024
#define G_ 4096          // 4*H
#define M_ (B_*T_)       // 16384

__device__ __forceinline__ float sigf(float x){ return 1.f/(1.f+__expf(-x)); }

// ---------------- elementwise converts / packs ----------------
__global__ void k_cvt_bf16(const float* __restrict__ in, __bf16* __restrict__ out, size_t n){
  size_t i = (size_t)blockIdx.x*blockDim.x + threadIdx.x;
  size_t stride = (size_t)gridDim.x*blockDim.x;
  for(; i<n; i+=stride) out[i] = (__bf16)in[i];
}

// W is [N,K] row-major fp32 (PyTorch weight). We need B[k,n] = W[n,k] fed to WMMA.
// Pack per 32x16 (KxN) tile so that lane L's v16bf fragment (K=(L/16)*16+e, N=L%16)
// is 32 contiguous bytes:  out[tile*512 + L*16 + e].
__global__ void k_pack_w(const float* __restrict__ W, __bf16* __restrict__ out, int N, int K){
  size_t gid = (size_t)blockIdx.x*blockDim.x + threadIdx.x;
  int ntiles = N >> 4;
  size_t total = (size_t)ntiles*(size_t)(K>>5)*32;
  if(gid >= total) return;
  int L = (int)(gid & 31);
  size_t tile = gid >> 5;
  int kt = (int)(tile / ntiles);
  int nt = (int)(tile % ntiles);
  const float* src = W + (size_t)(nt*16 + (L&15))*K + (size_t)kt*32 + (L>>4)*16;
  __bf16* dst = out + tile*512 + (size_t)L*16;
#pragma unroll
  for(int e=0;e<16;++e) dst[e] = (__bf16)src[e];
}

__global__ void k_add_bias(const float* __restrict__ a, const float* __restrict__ b,
                           float* __restrict__ o, int n){
  int i = blockIdx.x*blockDim.x + threadIdx.x;
  if(i<n) o[i] = a[i] + b[i];
}

// ---------------- bf16 WMMA GEMM: C[M,N] = A[M,K] * Bpack + bias + Cadd ----------------
// grid = (N/256, M/64), 256 threads = 8 waves in 2(M) x 4(N).
// Wave tile 32x64: 2 A fragments x 4 B fragments -> 8 WMMAs per 12 b128 loads.
__global__ __launch_bounds__(256) void k_wmma_gemm(
    const __bf16* __restrict__ A, const __bf16* __restrict__ Bp,
    const float* __restrict__ bias, const float* __restrict__ Cadd,
    long cadd_stride, float* __restrict__ C, long ldc, int K, int ntiles)
{
  const int lane = threadIdx.x & 31;
  const int wave = threadIdx.x >> 5;
  const int m0 = blockIdx.y*64 + (wave & 1)*32;
  const int n0 = blockIdx.x*256 + (wave >> 1)*64;
  const int half = lane >> 4;
  const int nt0 = n0 >> 4;
  const __bf16* arow0 = A + (size_t)(m0      + (lane & 15))*K;
  const __bf16* arow1 = A + (size_t)(m0 + 16 + (lane & 15))*K;

  v8f acc[2][4] = {};
  for(int kc=0; kc<K; kc+=32){
    // A fragments 16x32: lane half h holds K = 16h..16h+15 (contiguous, matches B pack).
    v8bf lo0 = *(const v8bf*)(arow0 + kc + half*16);
    v8bf hi0 = *(const v8bf*)(arow0 + kc + half*16 + 8);
    v8bf lo1 = *(const v8bf*)(arow1 + kc + half*16);
    v8bf hi1 = *(const v8bf*)(arow1 + kc + half*16 + 8);
    v16bf a0, a1;
#pragma unroll
    for(int j=0;j<8;++j){ a0[j]=lo0[j]; a0[j+8]=hi0[j]; a1[j]=lo1[j]; a1[j+8]=hi1[j]; }

    const __bf16* bbase = Bp + (((size_t)(kc>>5)*ntiles + nt0) << 9) + (size_t)lane*16;
    __builtin_prefetch(arow0 + kc + 32, 0, 3);                // global_prefetch next A chunks
    __builtin_prefetch(arow1 + kc + 32, 0, 3);
    __builtin_prefetch(bbase + ((size_t)ntiles << 9), 0, 3);  // next K-tile of B
#pragma unroll
    for(int i=0;i<4;++i){
      v16bf b = *(const v16bf*)(bbase + (i<<9));              // 32B contiguous fragment
      acc[0][i] = __builtin_amdgcn_wmma_f32_16x16x32_bf16(
                    false, a0, false, b, (short)0, acc[0][i], false, false);
      acc[1][i] = __builtin_amdgcn_wmma_f32_16x16x32_bf16(
                    false, a1, false, b, (short)0, acc[1][i], false, false);
    }
  }
  // C layout: lane L, VGPR r -> M = m0 + p*16 + (L/16)*8 + r, N = n0 + i*16 + L%16
  const int nb = n0 + (lane & 15);
#pragma unroll
  for(int p=0;p<2;++p){
    const int mb = m0 + p*16 + half*8;
#pragma unroll
    for(int i=0;i<4;++i){
      int n = nb + i*16;
      float badd = bias ? bias[n] : 0.f;
#pragma unroll
      for(int r=0;r<8;++r){
        int m = mb + r;
        float v = acc[p][i][r] + badd;
        if(Cadd) v += Cadd[(size_t)m*cadd_stride + n];
        C[(size_t)m*ldc + n] = v;
      }
    }
  }
}

// ---------------- LSTM gate pointwise ----------------
__global__ void k_lstm_pointwise(const float* __restrict__ g, float* __restrict__ c,
    float* __restrict__ hfp, __bf16* __restrict__ hbf, float* __restrict__ enc, int t){
  int gid = blockIdx.x*blockDim.x + threadIdx.x;
  if(gid >= B_*H_) return;
  int b = gid >> 10, h = gid & (H_-1);
  size_t base = (size_t)b*G_;
  float gi = g[base + h];
  float gf = g[base + H_   + h];
  float gg = g[base + 2*H_ + h];
  float go = g[base + 3*H_ + h];
  float cn = sigf(gf)*c[gid] + sigf(gi)*tanhf(gg);
  float hn = sigf(go)*tanhf(cn);
  c[gid]   = cn;
  hfp[gid] = hn;
  hbf[gid] = (__bf16)hn;
  if(enc) enc[((size_t)b*T_ + t)*H_ + h] = hn;
}

// ---------------- DAM covariances (x_shift == x with batch slots 31,63 zeroed) --------
__global__ void k_cov_temp(const float* __restrict__ x, float* __restrict__ covt){
  int lane = threadIdx.x & 31;
  int wid = (blockIdx.x*blockDim.x + threadIdx.x) >> 5;   // one wave per (b,t)
  if(wid >= B_*T_) return;
  int b = wid >> 8;
  bool zb = ((b & 31) == 31);
  const float* row = x + (size_t)wid*F_;
  float sx=0.f, sxs=0.f, sxx=0.f;
  for(int f=lane; f<F_; f+=32){
    float v  = row[f];
    float vs = zb ? 0.f : v;
    sx += v; sxs += vs; sxx += v*vs;
  }
#pragma unroll
  for(int o=16;o>0;o>>=1){
    sx  += __shfl_xor(sx, o, 32);
    sxs += __shfl_xor(sxs, o, 32);
    sxx += __shfl_xor(sxx, o, 32);
  }
  if(lane==0){
    float cov = (sxx - sx*sxs*(1.f/(float)F_)) * (1.f/(float)T_);
    covt[wid] = fmaxf(cov, 0.f);
  }
}

__global__ void k_cov_chan(const float* __restrict__ x, float* __restrict__ covc){
  int gid = blockIdx.x*blockDim.x + threadIdx.x;
  if(gid >= B_*F_) return;
  int b = gid >> 10, f = gid & (F_-1);
  bool zb = ((b & 31) == 31);
  float sx=0.f, sxs=0.f, sxx=0.f;
  for(int t=0;t<T_;++t){
    float v  = x[((size_t)b*T_ + t)*F_ + f];
    float vs = zb ? 0.f : v;
    sx += v; sxs += vs; sxx += v*vs;
  }
  float cov = (sxx - sx*sxs*(1.f/(float)T_)) * (1.f/(float)F_);
  covc[gid] = fmaxf(cov, 0.f);
}

__global__ void k_dist(const float* __restrict__ covt, const float* __restrict__ covc,
                       __bf16* __restrict__ out){
  size_t i = (size_t)blockIdx.x*blockDim.x + threadIdx.x;
  size_t n = (size_t)M_*F_;
  size_t stride = (size_t)gridDim.x*blockDim.x;
  for(; i<n; i+=stride){
    size_t bt = i >> 10;
    int f = (int)(i & (F_-1));
    int b = (int)(bt >> 8);
    out[i] = (__bf16)(covt[bt]*covc[b*F_ + f]);
  }
}

// ---------------- softmax over T per (b,h) ----------------
__global__ void k_softmax_stats(const float* __restrict__ enc,
                                float* __restrict__ smax, float* __restrict__ ssum){
  int gid = blockIdx.x*blockDim.x + threadIdx.x;
  if(gid >= B_*H_) return;
  int b = gid >> 10, h = gid & (H_-1);
  size_t base = (size_t)b*T_*H_ + h;
  float m = -1e30f;
  for(int t=0;t<T_;++t) m = fmaxf(m, enc[base + (size_t)t*H_]);
  float s = 0.f;
  for(int t=0;t<T_;++t) s += __expf(enc[base + (size_t)t*H_] - m);
  smax[gid] = m; ssum[gid] = s;
}

// ---------------- fused attention + MLP epilogue: one block per (b,t) ----------------
__global__ __launch_bounds__(64) void k_final(const float* __restrict__ enc,
    const float* __restrict__ smax, const float* __restrict__ ssum,
    const float* __restrict__ hlast, const float* __restrict__ w_attn,
    const float* __restrict__ b_attn, const float* __restrict__ w1,
    const float* __restrict__ b1, const float* __restrict__ w2,
    const float* __restrict__ b2, float* __restrict__ out){
  __shared__ float s_saa[H_];
  __shared__ float s_ta[64];
  __shared__ float s_r2[64];
  int bt = blockIdx.x;
  int b = bt >> 8;
  int tid = threadIdx.x;
  size_t ebase = (size_t)bt*H_;
  float ta = 0.f;
  for(int h=tid; h<H_; h+=64){
    float e = enc[ebase + h];
    int bh = b*H_ + h;
    float sp = __expf(e - smax[bh]) / ssum[bh];        // softmax over T
    s_saa[h] = hlast[bh]*(sp + 1.f);                   // saa = out_lstm*(attn+1)
    ta += tanhf(e)*w_attn[h];
  }
  s_ta[tid] = ta;
  __syncthreads();
  float s = b1[tid];
  for(int h=0; h<H_; ++h) s += s_saa[h]*w1[h*64 + tid];
  float r = fmaxf(s, 0.f);
  s_r2[tid] = r*w2[tid];
  __syncthreads();
  if(tid==0){
    float tsum=0.f, dsum=0.f;
#pragma unroll
    for(int j=0;j<64;++j){ tsum += s_ta[j]; dsum += s_r2[j]; }
    float temporal = sigf(tsum + b_attn[0]);
    float det = sigf(dsum + b2[0]);
    out[bt] = det + det*temporal;
  }
}

// ---------------- host orchestration ----------------
extern "C" void kernel_launch(void* const* d_in, const int* in_sizes, int n_in,
                              void* d_out, int out_size, void* d_ws, size_t ws_size,
                              hipStream_t stream){
  const float* x    = (const float*)d_in[0];
  const float* Wih1 = (const float*)d_in[1];
  const float* Whh1 = (const float*)d_in[2];
  const float* bih1 = (const float*)d_in[3];
  const float* bhh1 = (const float*)d_in[4];
  const float* Wih2 = (const float*)d_in[5];
  const float* Whh2 = (const float*)d_in[6];
  const float* bih2 = (const float*)d_in[7];
  const float* bhh2 = (const float*)d_in[8];
  const float* wat  = (const float*)d_in[9];
  const float* bat  = (const float*)d_in[10];
  const float* w1   = (const float*)d_in[11];
  const float* b1   = (const float*)d_in[12];
  const float* w2   = (const float*)d_in[13];
  const float* b2   = (const float*)d_in[14];
  float* out = (float*)d_out;
  (void)in_sizes; (void)n_in; (void)out_size; (void)ws_size;

  char* ws = (char*)d_ws;
  size_t off = 0;
  auto alloc = [&](size_t bytes)->char*{
    char* p = ws + off;
    off = (off + bytes + 255) & ~(size_t)255;
    return p;
  };
  __bf16* xbf  = (__bf16*)alloc((size_t)M_*F_*2);   // 32 MB
  __bf16* dbf  = (__bf16*)alloc((size_t)M_*F_*2);   // 32 MB
  __bf16* w1p  = (__bf16*)alloc((size_t)F_*G_*2);   // 8 MB each
  __bf16* wh1p = (__bf16*)alloc((size_t)H_*G_*2);
  __bf16* w2p  = (__bf16*)alloc((size_t)F_*G_*2);
  __bf16* wh2p = (__bf16*)alloc((size_t)H_*G_*2);
  float*  xg   = (float*)alloc((size_t)M_*G_*4);    // 256 MB (reused LSTM1/LSTM2)
  float*  encb = (float*)alloc((size_t)M_*H_*4);    // 64 MB
  float*  gbuf = (float*)alloc((size_t)B_*G_*4);
  float*  cbuf = (float*)alloc((size_t)B_*H_*4);
  float*  hfp1 = (float*)alloc((size_t)B_*H_*4);
  float*  hfp2 = (float*)alloc((size_t)B_*H_*4);
  __bf16* hbf  = (__bf16*)alloc((size_t)B_*H_*2);
  float*  covt = (float*)alloc((size_t)B_*T_*4);
  float*  covc = (float*)alloc((size_t)B_*F_*4);
  float*  bs1  = (float*)alloc((size_t)G_*4);
  float*  bs2  = (float*)alloc((size_t)G_*4);
  float*  smax = (float*)alloc((size_t)B_*H_*4);
  float*  ssum = (float*)alloc((size_t)B_*H_*4);

  // x -> bf16
  k_cvt_bf16<<<4096,256,0,stream>>>(x, xbf, (size_t)M_*F_);
  // pack weights to WMMA-fragment-native layout
  int pack_threads = (F_>>5)*(G_>>4)*32;   // 262144
  k_pack_w<<<pack_threads/256,256,0,stream>>>(Wih1, w1p,  G_, F_);
  k_pack_w<<<pack_threads/256,256,0,stream>>>(Whh1, wh1p, G_, H_);
  k_pack_w<<<pack_threads/256,256,0,stream>>>(Wih2, w2p,  G_, F_);
  k_pack_w<<<pack_threads/256,256,0,stream>>>(Whh2, wh2p, G_, H_);
  k_add_bias<<<16,256,0,stream>>>(bih1, bhh1, bs1, G_);
  k_add_bias<<<16,256,0,stream>>>(bih2, bhh2, bs2, G_);

  dim3 ggrid(G_/256, M_/64);   // big input GEMM
  dim3 sgrid(G_/256, B_/64);   // per-step recurrent GEMM

  // xg1 = x @ W_ih1^T + (b_ih1 + b_hh1)
  k_wmma_gemm<<<ggrid,256,0,stream>>>(xbf, w1p, bs1, nullptr, 0, xg, G_, F_, G_>>4);

  // LSTM 1 (need only final h -> hfp1)
  hipMemsetAsync(cbuf, 0, (size_t)B_*H_*4, stream);
  hipMemsetAsync(hbf,  0, (size_t)B_*H_*2, stream);
  for(int t=0;t<T_;++t){
    k_wmma_gemm<<<sgrid,256,0,stream>>>(hbf, wh1p, nullptr,
        xg + (size_t)t*G_, (long)T_*G_, gbuf, G_, H_, G_>>4);
    k_lstm_pointwise<<<(B_*H_)/256,256,0,stream>>>(gbuf, cbuf, hfp1, hbf, nullptr, t);
  }

  // DAM covariances and dist (bf16 for GEMM 2)
  k_cov_temp<<<(B_*T_*32)/256,256,0,stream>>>(x, covt);
  k_cov_chan<<<(B_*F_)/256,256,0,stream>>>(x, covc);
  k_dist<<<4096,256,0,stream>>>(covt, covc, dbf);

  // xg2 = dist @ W_ih2^T + (b_ih2 + b_hh2)   (reuses xg buffer)
  k_wmma_gemm<<<ggrid,256,0,stream>>>(dbf, w2p, bs2, nullptr, 0, xg, G_, F_, G_>>4);

  // LSTM 2 (full hidden sequence -> encb)
  hipMemsetAsync(cbuf, 0, (size_t)B_*H_*4, stream);
  hipMemsetAsync(hbf,  0, (size_t)B_*H_*2, stream);
  for(int t=0;t<T_;++t){
    k_wmma_gemm<<<sgrid,256,0,stream>>>(hbf, wh2p, nullptr,
        xg + (size_t)t*G_, (long)T_*G_, gbuf, G_, H_, G_>>4);
    k_lstm_pointwise<<<(B_*H_)/256,256,0,stream>>>(gbuf, cbuf, hfp2, hbf, encb, t);
  }

  // softmax stats + fused attention/MLP epilogue
  k_softmax_stats<<<(B_*H_)/256,256,0,stream>>>(encb, smax, ssum);
  k_final<<<B_*T_,64,0,stream>>>(encb, smax, ssum, hfp1, wat, bat, w1, b1, w2, b2, out);
}
